// GRU_53549652247344
// MI455X (gfx1250) — matrix operand order repaired and checked by reading, not compiled, not run on hardware
//
#include <hip/hip_runtime.h>
#include <hip/hip_bf16.h>

typedef __attribute__((ext_vector_type(16))) __bf16 v16bf;
typedef __attribute__((ext_vector_type(8)))  __bf16 v8bf;
typedef __attribute__((ext_vector_type(8)))  float  v8f;

#define B_ 64
#define T_ 512
#define I_ 512
#define H_ 1024
#define O_ 512
#define KTOT 1536            // I_ + H_
#define NKC 48               // KTOT / 32
#define A_STRIDE 1544        // KTOT + 8 bf16 pad (keeps 16B alignment, staggers banks)
#define NSLICE 64            // H_/16 workgroups, one 16-col slice each
#define WSLICE (NKC*512)     // 24576 bf16 per matrix slice (swizzled B fragments)
#define NSW ((size_t)KTOT*H_)// 1572864 bf16 per full matrix

// workspace layout (bytes)
static const size_t OFF_CNT  = 0;                                   // 16 B counter
static const size_t OFF_HBUF = 256;                                 // 2 * 64*1024 bf16 = 262144
static const size_t OFF_HFIN = OFF_HBUF + 2ull*B_*H_*2;             // 64*1024 f32  = 262144
static const size_t OFF_XB   = OFF_HFIN + (size_t)B_*H_*4;          // T*B*I bf16   = 33554432
static const size_t OFF_WSW  = OFF_XB + (size_t)T_*B_*I_*2;         // 2*NSW bf16   = 6291456

__global__ void k_init(unsigned* p, int n) {
  for (int i = blockIdx.x * blockDim.x + threadIdx.x; i < n; i += gridDim.x * blockDim.x)
    p[i] = 0u;
}

// x [B,T,I] f32 -> xb [T,B,I] bf16 (contiguous per-timestep tiles)
__global__ void k_convx(const float* __restrict__ x, __bf16* __restrict__ xb) {
  const size_t n = (size_t)B_ * T_ * I_;
  for (size_t idx = (size_t)blockIdx.x * blockDim.x + threadIdx.x; idx < n;
       idx += (size_t)gridDim.x * blockDim.x) {
    size_t b = idx / ((size_t)T_ * I_);
    size_t rem = idx % ((size_t)T_ * I_);
    size_t t = rem / I_, i = rem % I_;
    xb[(t * B_ + b) * I_ + i] = (__bf16)x[idx];
  }
}

// W_z/W_h [1536,1024] f32 -> bf16 swizzled into per-(slice,kchunk,lane) B fragments
__global__ void k_convw(const float* __restrict__ Wz, const float* __restrict__ Wh,
                        __bf16* __restrict__ wsw) {
  const size_t n = 2ull * NSW;
  for (size_t idx = (size_t)blockIdx.x * blockDim.x + threadIdx.x; idx < n;
       idx += (size_t)gridDim.x * blockDim.x) {
    int mat = idx >= NSW;
    size_t r = idx - (mat ? NSW : 0);
    int s  = (int)(r / WSLICE);
    int r2 = (int)(r % WSLICE);
    int kc = r2 >> 9;
    int r3 = r2 & 511;
    int l = r3 >> 4, j = r3 & 15;
    // B 32x16 bf16 layout: lanes 0-15 hold K {0..7,16..23}, lanes 16-31 hold K {8..15,24..31}, N = lane&15
    int K = kc * 32 + ((j >> 3) << 4) + ((l >> 4) << 3) + (j & 7);
    int nn = s * 16 + (l & 15);
    const float* src = mat ? Wh : Wz;
    wsw[idx] = (__bf16)src[(size_t)K * H_ + nn];
  }
}

// ---- CDNA5 data movers ----------------------------------------------------

#if __has_builtin(__builtin_amdgcn_tensor_load_to_lds)
#define HAVE_TDM 1
typedef __attribute__((ext_vector_type(4))) unsigned int v4u;
typedef __attribute__((ext_vector_type(8))) int v8i;
typedef __attribute__((ext_vector_type(4))) int v4i;

// Linear TDM copy: nbytes (8B-aligned, nbytes/8 <= 65535) global -> LDS.
// 1D tensor expressed as a 2D tile [u x 1], data_size = 8B, D#.type = 2.
__device__ __forceinline__ void tdm_load_lin(unsigned lds_addr, const void* g, unsigned nbytes) {
  unsigned long long ga = (unsigned long long)(size_t)g;
  unsigned u = nbytes >> 3;                              // 8-byte units
  v4u g0;
  g0.x = 1u;                                             // count=1 valid descriptor
  g0.y = lds_addr;                                       // lds_addr [63:32]
  g0.z = (unsigned)ga;                                   // global_addr [95:64]
  g0.w = (unsigned)((ga >> 32) & 0x01FFFFFFull) | 0x80000000u; // addr[56:32] | type=2
  v8i g1;
  g1[0] = (int)(3u << 16);                               // data_size = 8B
  g1[1] = (int)((u & 0xFFFFu) << 16);                    // tensor_dim0 [15:0] @ bits63:48
  g1[2] = (int)(((u >> 16) & 0xFFFFu) | (1u << 16));     // tensor_dim0 hi | tensor_dim1 = 1
  g1[3] = (int)(u << 16);                                // tile_dim0 = u
  g1[4] = 1;                                             // tile_dim1 = 1, tile_dim2 = 0
  g1[5] = (int)u;                                        // tensor_dim0_stride lo
  g1[6] = 0;
  g1[7] = 0;
  v4i z4 = {0, 0, 0, 0};
#if defined(__clang_major__) && __clang_major__ >= 23
  v8i z8 = {0, 0, 0, 0, 0, 0, 0, 0};
  __builtin_amdgcn_tensor_load_to_lds(g0, g1, z4, z4, z8, 0);
#else
  __builtin_amdgcn_tensor_load_to_lds(g0, g1, z4, z4, 0);
#endif
}
#endif

// Per-lane async global->LDS 16B copy (GVS mode), tracked by ASYNCcnt.
__device__ __forceinline__ void async_ld16(unsigned lds_addr, unsigned long long sbase,
                                           unsigned voff) {
  asm volatile("global_load_async_to_lds_b128 %0, %1, %2"
               :: "v"(lds_addr), "v"(voff), "s"(sbase) : "memory");
}
__device__ __forceinline__ void wait_async0() {
  asm volatile("s_wait_asynccnt 0x0" ::: "memory");
}

// ---------------------------------------------------------------------------

__global__ void __launch_bounds__(256, 1)
k_gru(const __bf16* __restrict__ xb, const __bf16* __restrict__ wsw,
      const float* __restrict__ bz, const float* __restrict__ bh,
      const float* __restrict__ fcw, const float* __restrict__ fcb,
      __bf16* hbuf, float* hfin, unsigned* cnt, float* out) {
  extern __shared__ char smem[];
  __bf16* comb = (__bf16*)smem;                 // [64][A_STRIDE] bf16 activations
  __bf16* ldsw = comb + 64 * A_STRIDE;          // [2][WSLICE]   bf16 weight fragments
  float*  accf = (float*)(ldsw + 2 * WSLICE);   // [2][64][16]   f32 gate accumulators
  float*  hprev = accf + 2 * 64 * 16;           // [64][16]      f32 local h slice
  float*  biasz = hprev + 64 * 16;
  float*  biash = biasz + 16;

  const int tid = threadIdx.x;
  const int lane = tid & 31;
  const int wave = tid >> 5;
  const int slice = blockIdx.x;     // owns H columns [n0, n0+16)
  const int n0 = slice * 16;
  const unsigned nwg = gridDim.x;

  // LDS byte offsets (LDS aperture: generic addr[31:0] == LDS byte offset)
  const unsigned comb_off = (unsigned)(size_t)comb;
  const unsigned ldsw_off = (unsigned)(size_t)ldsw;

  // ---- prologue: weights -> LDS once via Tensor Data Mover ----
#if defined(HAVE_TDM)
  if (wave == 0) {
    tdm_load_lin(ldsw_off,              wsw + (size_t)slice * WSLICE,       WSLICE * 2);
    tdm_load_lin(ldsw_off + WSLICE * 2, wsw + NSW + (size_t)slice * WSLICE, WSLICE * 2);
#if __has_builtin(__builtin_amdgcn_s_wait_tensorcnt)
    __builtin_amdgcn_s_wait_tensorcnt(0);
#else
    asm volatile("s_wait_tensorcnt 0x0" ::: "memory");
#endif
  }
#else
  {
    const uint4* g0 = (const uint4*)(wsw + (size_t)slice * WSLICE);
    const uint4* g1 = (const uint4*)(wsw + NSW + (size_t)slice * WSLICE);
    uint4* d0 = (uint4*)ldsw;
    uint4* d1 = (uint4*)(ldsw + WSLICE);
    for (int i = tid; i < WSLICE / 8; i += 256) { d0[i] = g0[i]; d1[i] = g1[i]; }
  }
#endif
  for (int e = tid; e < 64 * 16; e += 256) hprev[e] = 0.f;
  if (tid < 16) { biasz[tid] = bz[n0 + tid]; biash[tid] = bh[n0 + tid]; }
  __syncthreads();

  const int mat = wave >> 2;      // 0: z-gate (W_z), 1: candidate (W_h)
  const int mtile = wave & 3;     // M tile (rows 16*mtile .. +15)
  // A 16x32 bf16 layout: lane<16 -> K {0..7,16..23}; lane>=16 -> K {8..15,24..31}
  const __bf16* abase = comb + (mtile * 16 + (lane & 15)) * A_STRIDE + ((lane & 16) ? 8 : 0);
  const __bf16* bbase = ldsw + mat * WSLICE + lane * 16;

  unsigned epoch = 0;
  for (int t = 0; t < T_; ++t) {
    const int p = t & 1;
    const __bf16* hread = hbuf + p * (B_ * H_);
    __bf16* hwrite = hbuf + (p ^ 1) * (B_ * H_);

    // stage combined = [x_t | h] into LDS via async global->LDS (no VGPR round trip)
    {
      unsigned long long xs = (unsigned long long)(size_t)(xb + (size_t)t * (B_ * I_));
      for (int idx = tid; idx < 64 * 64; idx += 256) {
        int row = idx >> 6, seg = idx & 63;
        async_ld16(comb_off + row * (A_STRIDE * 2) + seg * 16, xs, (unsigned)(idx * 16));
      }
      unsigned long long hs = (unsigned long long)(size_t)hread;
      for (int idx = tid; idx < 64 * 128; idx += 256) {
        int row = idx >> 7, seg = idx & 127;
        async_ld16(comb_off + row * (A_STRIDE * 2) + I_ * 2 + seg * 16, hs,
                   (unsigned)(idx * 16));
      }
    }
    if (t + 1 < T_)
      __builtin_prefetch((const void*)(xb + (size_t)(t + 1) * (B_ * I_) + tid * 128), 0, 0);
    wait_async0();
    __syncthreads();

    // [64,1536] x [1536,16] for this wave's gate/M-tile: 48 bf16 WMMAs
    v8f c = {0.f, 0.f, 0.f, 0.f, 0.f, 0.f, 0.f, 0.f};
#pragma unroll 4
    for (int kc = 0; kc < NKC; ++kc) {
      v8bf alo = *(const v8bf*)(abase + kc * 32);
      v8bf ahi = *(const v8bf*)(abase + kc * 32 + 16);
      v8bf blo = *(const v8bf*)(bbase + kc * 512);
      v8bf bhi = *(const v8bf*)(bbase + kc * 512 + 8);
      v16bf a = __builtin_shufflevector(alo, ahi, 0,1,2,3,4,5,6,7,8,9,10,11,12,13,14,15);
      v16bf b = __builtin_shufflevector(blo, bhi, 0,1,2,3,4,5,6,7,8,9,10,11,12,13,14,15);
      c = __builtin_amdgcn_wmma_f32_16x16x32_bf16(false, a, false, b, (short)0, c, false, false);
    }

    // spill accumulator tile to LDS (C layout: VGPR j -> M = j (+8 for hi lanes), N = lane&15)
    {
      float* accp = accf + mat * (64 * 16);
      int crow = mtile * 16 + ((lane & 16) ? 8 : 0);
      int ccol = lane & 15;
#pragma unroll
      for (int j = 0; j < 8; ++j) accp[(crow + j) * 16 + ccol] = c[j];
    }
    __syncthreads();

    // elementwise gate math (fast sigmoid/tanh: v_exp + v_rcp, no IEEE divide)
    for (int e = tid; e < 64 * 16; e += 256) {
      int row = e >> 4, col = e & 15;
      float za = accf[e] + biasz[col];
      float ha = accf[64 * 16 + e] + biash[col];
      float z = __builtin_amdgcn_rcpf(1.f + __expf(-za));
      float th = 2.f * __builtin_amdgcn_rcpf(1.f + __expf(-2.f * ha)) - 1.f;
      float hp = hprev[e];
      float hn = (1.f - z) * hp + z * th;
      hprev[e] = hn;
      hwrite[row * H_ + n0 + col] = (__bf16)hn;
    }
    __threadfence();

    // grid-wide barrier between timesteps (monotonic counter)
    __syncthreads();
    if (tid == 0) {
      __hip_atomic_fetch_add(cnt, 1u, __ATOMIC_ACQ_REL, __HIP_MEMORY_SCOPE_AGENT);
      ++epoch;
      unsigned target = epoch * nwg;
      while (__hip_atomic_load(cnt, __ATOMIC_ACQUIRE, __HIP_MEMORY_SCOPE_AGENT) < target)
        __builtin_amdgcn_s_sleep(2);
    }
    __syncthreads();
  }

  // publish final h slice in f32, barrier, then FC tail
  for (int e = tid; e < 64 * 16; e += 256) {
    int row = e >> 4, col = e & 15;
    hfin[row * H_ + n0 + col] = hprev[e];
  }
  __threadfence();
  __syncthreads();
  if (tid == 0) {
    __hip_atomic_fetch_add(cnt, 1u, __ATOMIC_ACQ_REL, __HIP_MEMORY_SCOPE_AGENT);
    ++epoch;
    unsigned target = epoch * nwg;
    while (__hip_atomic_load(cnt, __ATOMIC_ACQUIRE, __HIP_MEMORY_SCOPE_AGENT) < target)
      __builtin_amdgcn_s_sleep(2);
  }
  __syncthreads();

  // y = h_final @ fc_w^T + fc_b : each WG computes 8 output columns, f32
  const int o0 = blockIdx.x * (O_ / NSLICE);
  for (int q = tid; q < B_ * (O_ / NSLICE); q += 256) {
    int b = q >> 3;
    int o = o0 + (q & 7);
    const float4* hr = (const float4*)(hfin + (size_t)b * H_);
    const float4* wr = (const float4*)(fcw + (size_t)o * H_);
    float s = fcb[o];
    for (int k = 0; k < H_ / 4; ++k) {
      float4 hv = hr[k], wv = wr[k];
      s += hv.x * wv.x + hv.y * wv.y + hv.z * wv.z + hv.w * wv.w;
    }
    out[b * O_ + o] = s;
  }
}

extern "C" void kernel_launch(void* const* d_in, const int* in_sizes, int n_in,
                              void* d_out, int out_size, void* d_ws, size_t ws_size,
                              hipStream_t stream) {
  (void)in_sizes; (void)n_in; (void)out_size; (void)ws_size;
  const float* x   = (const float*)d_in[0];
  // d_in[1]=W_r, d_in[2]=b_r are dead in the reference (r is unused)
  const float* W_z = (const float*)d_in[3];
  const float* b_z = (const float*)d_in[4];
  const float* W_h = (const float*)d_in[5];
  const float* b_h = (const float*)d_in[6];
  const float* fcw = (const float*)d_in[7];
  const float* fcb = (const float*)d_in[8];
  float* out = (float*)d_out;

  char* ws = (char*)d_ws;
  unsigned* cnt = (unsigned*)(ws + OFF_CNT);
  __bf16* hbuf  = (__bf16*)(ws + OFF_HBUF);
  float*  hfin  = (float*)(ws + OFF_HFIN);
  __bf16* xb    = (__bf16*)(ws + OFF_XB);
  __bf16* wsw   = (__bf16*)(ws + OFF_WSW);

  // zero counter + h double-buffer each call (deterministic under graph replay)
  k_init<<<256, 256, 0, stream>>>((unsigned*)ws, (int)(OFF_HFIN / 4));
  k_convx<<<4096, 256, 0, stream>>>(x, xb);
  k_convw<<<2048, 256, 0, stream>>>(W_z, W_h, wsw);

  size_t smem = (size_t)(64 * A_STRIDE + 2 * WSLICE) * sizeof(__bf16) +
                (size_t)(2 * 64 * 16 + 64 * 16 + 32) * sizeof(float);   // 308352 B
  k_gru<<<NSLICE, 256, smem, stream>>>(xb, wsw, b_z, b_h, fcw, fcb, hbuf, hfin, cnt, out);
}
